// Vector_Quantizer_47107201302664
// MI455X (gfx1250) — matrix operand order repaired and checked by reading, not compiled, not run on hardware
//
#include <hip/hip_runtime.h>
#include <hip/hip_bf16.h>
#include <stdint.h>

// ---------------------------------------------------------------------------
// Problem sizes (fixed by the reference)
// ---------------------------------------------------------------------------
#define BB   1024
#define LL   64
#define EE   1024
#define DCC  768
#define NN   16384
#define EPSF 1e-8f

typedef __attribute__((ext_vector_type(16))) __bf16 bf16x16;
typedef __attribute__((ext_vector_type(8)))  float  f32x8;

union Frag16 { bf16x16 v; uint32_t u[8]; };
union FragC  { f32x8   v; float    f[8]; };

__device__ __forceinline__ uint16_t f32_to_bf16_bits(float f) {
    uint32_t u = __float_as_uint(f);
    uint32_t r = u + 0x7FFFu + ((u >> 16) & 1u);   // round-to-nearest-even
    return (uint16_t)(r >> 16);
}

// ---------------------------------------------------------------------------
// Elementwise f32 -> bf16 conversion / f32 zero fill
// ---------------------------------------------------------------------------
__global__ void k_cvt_bf16(const float* __restrict__ in, uint16_t* __restrict__ out, int n) {
    int i = blockIdx.x * blockDim.x + threadIdx.x;
    if (i < n) out[i] = f32_to_bf16_bits(in[i]);
}

__global__ void k_zero_f32(float* __restrict__ p, int n) {
    int i = blockIdx.x * blockDim.x + threadIdx.x;
    if (i < n) p[i] = 0.f;
}

// ---------------------------------------------------------------------------
// Codebook prep: bf16 copy [N,D], bf16 transpose [D,N], per-row L2 norm.
// One 256-thread block per codebook row.
// ---------------------------------------------------------------------------
__global__ void k_codebook_prep(const float* __restrict__ W,
                                uint16_t* __restrict__ cb,   // [N,D] bf16
                                uint16_t* __restrict__ cbt,  // [D,N] bf16
                                float* __restrict__ cnorm,   // [N]
                                int N, int D) {
    int n = blockIdx.x;
    int tid = threadIdx.x;
    __shared__ float red[256];
    float ssq = 0.f;
    for (int d = tid; d < D; d += blockDim.x) {
        float w = W[(size_t)n * D + d];
        ssq += w * w;
        uint16_t bw = f32_to_bf16_bits(w);
        cb[(size_t)n * D + d] = bw;
        cbt[(size_t)d * N + n] = bw;
    }
    red[tid] = ssq; __syncthreads();
    for (int s = 128; s > 0; s >>= 1) { if (tid < s) red[tid] += red[tid + s]; __syncthreads(); }
    if (tid == 0) cnorm[n] = fmaxf(sqrtf(red[0]), EPSF);
}

// ---------------------------------------------------------------------------
// Blocked bf16 WMMA GEMM:  C[M,N] = A[M,K] * Bt[N,K]^T  (+ bias[N])
// Each wave computes an (MT*16) x (NT*16) strip: MT A-frags and NT B-frags
// feed MT*NT independent WMMA accumulators per K-step (no RAW hazards, high
// wmma:vmem ratio).  Optional K-split: partial products accumulated into a
// zero-initialized C via global_atomic_add_f32 (no-return).
// Fragment layouts per ISA 7.12.2 (16-bit A 16x32; B column-major pairs).
// ---------------------------------------------------------------------------
template <int MT, int NT>
__global__ void k_wmma_gemm(const uint16_t* __restrict__ A,
                            const uint16_t* __restrict__ Bt,
                            float* __restrict__ C,
                            const float* __restrict__ bias,
                            int M, int N, int K, int kSplit) {
    int wave   = blockIdx.x * (blockDim.x >> 5) + (threadIdx.x >> 5);
    int tilesN = N / (16 * NT);
    int tilesM = M / (16 * MT);
    if (wave >= tilesM * tilesN * kSplit) return;  // wave-uniform: EXEC stays all-ones

    int tk = wave / (tilesM * tilesN);
    int t  = wave % (tilesM * tilesN);
    int tm = t / tilesN;
    int tn = t % tilesN;

    int lane = threadIdx.x & 31;
    int m = lane & 15;
    int h = lane >> 4;

    int rowBase = tm * 16 * MT;
    int colBase = tn * 16 * NT;

    int Ksub = K / kSplit;
    int k0   = tk * Ksub;

    const uint16_t* Arow[MT];
    const uint16_t* Brow[NT];
#pragma unroll
    for (int mi = 0; mi < MT; ++mi) Arow[mi] = A  + (size_t)(rowBase + mi * 16 + m) * K + k0;
#pragma unroll
    for (int ni = 0; ni < NT; ++ni) Brow[ni] = Bt + (size_t)(colBase + ni * 16 + m) * K + k0;

    FragC acc[MT][NT];
#pragma unroll
    for (int mi = 0; mi < MT; ++mi)
#pragma unroll
        for (int ni = 0; ni < NT; ++ni)
#pragma unroll
            for (int i = 0; i < 8; ++i) acc[mi][ni].f[i] = 0.f;

    for (int kb = 0; kb < Ksub; kb += 32) {
        if (kb + 128 < Ksub) {
#pragma unroll
            for (int mi = 0; mi < MT; ++mi) __builtin_prefetch(Arow[mi] + kb + 128, 0, 1);
#pragma unroll
            for (int ni = 0; ni < NT; ++ni) __builtin_prefetch(Brow[ni] + kb + 128, 0, 1);
        }
        Frag16 a[MT], b[NT];
#pragma unroll
        for (int mi = 0; mi < MT; ++mi)
#pragma unroll
            for (int v = 0; v < 8; ++v) {
                // A (16x32 bf16): vgpr v holds K pair 2*(v&3) + 8*h + 16*(v>>2)
                int ka = kb + 2 * (v & 3) + 8 * h + ((v >> 2) << 4);
                a[mi].u[v] = *(const uint32_t*)(Arow[mi] + ka);
            }
#pragma unroll
        for (int ni = 0; ni < NT; ++ni)
#pragma unroll
            for (int v = 0; v < 8; ++v) {
                // B column-major (32x16): vgpr v holds K pair 2*(v&3) + 8*h + 16*(v>>2)
                int kbi = kb + 2 * (v & 3) + 8 * h + ((v >> 2) << 4);
                b[ni].u[v] = *(const uint32_t*)(Brow[ni] + kbi);
            }
#pragma unroll
        for (int mi = 0; mi < MT; ++mi)
#pragma unroll
            for (int ni = 0; ni < NT; ++ni)
                acc[mi][ni].v = __builtin_amdgcn_wmma_f32_16x16x32_bf16(
                    false, a[mi].v, false, b[ni].v,
                    (short)0, acc[mi][ni].v, false, false);
    }

    // D layout: lane(0-15): N=lane, M=r; lane(16-31): N=lane-16, M=8+r
#pragma unroll
    for (int mi = 0; mi < MT; ++mi)
#pragma unroll
        for (int ni = 0; ni < NT; ++ni) {
            int col = colBase + ni * 16 + m;
            if (kSplit == 1) {
                float bv = bias ? bias[col] : 0.f;
#pragma unroll
                for (int r = 0; r < 8; ++r) {
                    int row = rowBase + mi * 16 + r + 8 * h;
                    C[(size_t)row * N + col] = acc[mi][ni].f[r] + bv;
                }
            } else {
#pragma unroll
                for (int r = 0; r < 8; ++r) {
                    int row = rowBase + mi * 16 + r + 8 * h;
                    atomicAdd(&C[(size_t)row * N + col], acc[mi][ni].f[r]);
                }
            }
        }
}

// ---------------------------------------------------------------------------
// Per-row L2 norm of projected x + bf16 copy.  One block per batch row.
// ---------------------------------------------------------------------------
__global__ void k_xnorm(const float* __restrict__ xf, uint16_t* __restrict__ xb,
                        float* __restrict__ xnorm, int D) {
    int b = blockIdx.x;
    int tid = threadIdx.x;
    __shared__ float red[256];
    float ssq = 0.f;
    for (int d = tid; d < D; d += blockDim.x) {
        float v = xf[(size_t)b * D + d];
        ssq += v * v;
        xb[(size_t)b * D + d] = f32_to_bf16_bits(v);
    }
    red[tid] = ssq; __syncthreads();
    for (int s = 128; s > 0; s >>= 1) { if (tid < s) red[tid] += red[tid + s]; __syncthreads(); }
    if (tid == 0) xnorm[b] = fmaxf(sqrtf(red[0]), EPSF);
}

// ---------------------------------------------------------------------------
// Softmax over codebook axis.  logit = -S/(||x||*||W_n||)  (the "1 - cos"
// constant shift cancels in softmax).  Stores UNNORMALIZED exp in bf16;
// 1/sum is applied when the clue row is emitted.  One block per batch row.
// ---------------------------------------------------------------------------
__global__ void k_softmax(const float* __restrict__ S, const float* __restrict__ cnorm,
                          const float* __restrict__ xnorm, uint16_t* __restrict__ pb,
                          float* __restrict__ invsum, int N) {
    int b = blockIdx.x;
    int tid = threadIdx.x;
    __shared__ float red[256];
    float rxn = 1.f / xnorm[b];

    float mx = -1e30f;
    for (int n = tid; n < N; n += 256) {
        float l = -S[(size_t)b * N + n] * rxn / cnorm[n];
        mx = fmaxf(mx, l);
    }
    red[tid] = mx; __syncthreads();
    for (int s = 128; s > 0; s >>= 1) { if (tid < s) red[tid] = fmaxf(red[tid], red[tid + s]); __syncthreads(); }
    mx = red[0]; __syncthreads();

    float sum = 0.f;
    for (int n = tid; n < N; n += 256) {
        float l = -S[(size_t)b * N + n] * rxn / cnorm[n];
        float e = __expf(l - mx);
        sum += e;
        pb[(size_t)b * N + n] = f32_to_bf16_bits(e);
    }
    red[tid] = sum; __syncthreads();
    for (int s = 128; s > 0; s >>= 1) { if (tid < s) red[tid] += red[tid + s]; __syncthreads(); }
    if (tid == 0) invsum[b] = 1.f / red[0];
}

// ---------------------------------------------------------------------------
// Valid length per batch row: vl = L - sum(mask)
// ---------------------------------------------------------------------------
__global__ void k_vl(const int* __restrict__ mask, int* __restrict__ vl, int B, int L) {
    int b = blockIdx.x * blockDim.x + threadIdx.x;
    if (b < B) {
        int s = 0;
        for (int l = 0; l < L; ++l) s += (mask[b * L + l] != 0);
        vl[b] = L - s;
    }
}

// ---------------------------------------------------------------------------
// Assemble outputs.  One block per (b, l) row of 768 floats.
//   l == vl   -> text_emb[b, vl-1]          (shift-copy)
//   l == vl-1 -> clue[b] = U[b] * invsum[b] (softmax readout)
//   else      -> text_emb[b, l]
// mask out: attn_mask & ~(do & at_vl), as f32 0/1.  Block 0 writes scalar 0.
// ---------------------------------------------------------------------------
__global__ void k_emit(const float* __restrict__ text_emb, const int* __restrict__ mask,
                       const float* __restrict__ U, const float* __restrict__ invsum,
                       const int* __restrict__ vl, float* __restrict__ out_emb,
                       float* __restrict__ out_mask, float* __restrict__ out_scalar,
                       int L, int D) {
    int bl = blockIdx.x;
    int b = bl / L, l = bl % L;
    int v = vl[b];
    bool doins = v < L;

    const float* src;
    bool use_clue = false;
    float scale = 1.f;
    if (doins && l == v) {
        int p = v - 1; p = p < 0 ? 0 : (p > L - 1 ? L - 1 : p);
        src = text_emb + ((size_t)b * L + p) * D;
    } else if (doins && l == v - 1) {
        src = U + (size_t)b * D;
        use_clue = true;
        scale = invsum[b];
    } else {
        src = text_emb + ((size_t)b * L + l) * D;
    }
    for (int d = threadIdx.x; d < D; d += blockDim.x) {
        float val = src[d];
        if (use_clue) val *= scale;
        out_emb[((size_t)b * L + l) * D + d] = val;
    }
    if (threadIdx.x == 0) {
        int mv = (mask[b * L + l] != 0);
        int nm = mv && !(doins && l == v);
        out_mask[b * L + l] = (float)nm;
        if (bl == 0) out_scalar[0] = 0.0f;
    }
}

// ---------------------------------------------------------------------------
// Host launcher
// ---------------------------------------------------------------------------
extern "C" void kernel_launch(void* const* d_in, const int* in_sizes, int n_in,
                              void* d_out, int out_size, void* d_ws, size_t ws_size,
                              hipStream_t stream) {
    (void)in_sizes; (void)n_in; (void)out_size; (void)ws_size;

    const float* inputs   = (const float*)d_in[0];   // [B,1,E]
    const float* text_emb = (const float*)d_in[1];   // [B,L,DC]
    const int*   attn     = (const int*)  d_in[2];   // [B,L] bool->int
    const float* proj_w   = (const float*)d_in[3];   // [DC,E]
    const float* proj_b   = (const float*)d_in[4];   // [DC]
    const float* codebook = (const float*)d_in[5];   // [N,DC]

    // ---- workspace layout (bytes) ----
    char* w = (char*)d_ws;
    float*    S      = (float*)   (w + 0);                       // [B,N]   64 MB
    uint16_t* pb     = (uint16_t*)(w + 67108864);                // [B,N]   32 MB
    uint16_t* cb     = (uint16_t*)(w + 100663296);               // [N,DC]  24 MB
    uint16_t* cbt    = (uint16_t*)(w + 125829120);               // [DC,N]  24 MB
    uint16_t* ib     = (uint16_t*)(w + 150994944);               // [B,E]    2 MB
    uint16_t* wb     = (uint16_t*)(w + 153092096);               // [DC,E] 1.5 MB
    float*    xf     = (float*)   (w + 154664960);               // [B,DC]   3 MB
    uint16_t* xb     = (uint16_t*)(w + 157810688);               // [B,DC] 1.5 MB
    float*    U      = (float*)   (w + 159383552);               // [B,DC]   3 MB
    float*    cnorm  = (float*)   (w + 162529280);               // [N]
    float*    xnorm  = (float*)   (w + 162594816);               // [B]
    float*    invsum = (float*)   (w + 162598912);               // [B]
    int*      vl     = (int*)     (w + 162603008);               // [B]

    float* out_emb    = (float*)d_out;                     // B*L*DC
    float* out_mask   = out_emb + (size_t)BB * LL * DCC;   // B*L
    float* out_scalar = out_mask + (size_t)BB * LL;        // 1

    // 1) bf16 conversions of GEMM operands (+ zero K-split accumulator)
    k_cvt_bf16<<<(BB * EE) / 256, 256, 0, stream>>>(inputs, ib, BB * EE);
    k_cvt_bf16<<<(DCC * EE) / 256, 256, 0, stream>>>(proj_w, wb, DCC * EE);
    k_codebook_prep<<<NN, 256, 0, stream>>>(codebook, cb, cbt, cnorm, NN, DCC);
    k_zero_f32<<<(BB * DCC) / 256, 256, 0, stream>>>(U, BB * DCC);

    // 2) x = inputs @ proj_w^T + b   (M=1024, N=768, K=1024)
    //    32x64 strips: (1024/32)*(768/64) = 384 waves -> 48 blocks
    k_wmma_gemm<2, 4><<<(BB / 32) * (DCC / 64) / 8, 256, 0, stream>>>(
        ib, wb, xf, proj_b, BB, DCC, EE, 1);

    // 3) per-row norm of x, bf16 copy
    k_xnorm<<<BB, 256, 0, stream>>>(xf, xb, xnorm, DCC);

    // 4) S = x @ codebook^T  (M=1024, N=16384, K=768)
    //    (1024/32)*(16384/64) = 8192 waves -> 1024 blocks
    k_wmma_gemm<2, 4><<<(BB / 32) * (NN / 64) / 8, 256, 0, stream>>>(
        xb, cb, S, nullptr, BB, NN, DCC, 1);

    // 5) softmax over N -> unnormalized exp (bf16) + 1/sum
    k_softmax<<<BB, 256, 0, stream>>>(S, cnorm, xnorm, pb, invsum, NN);

    // 6) U = exp @ codebook  (M=1024, N=768, K=16384)
    //    K-split x8 for load balance: 384*8 = 3072 waves -> 384 blocks,
    //    partials merged with global_atomic_add_f32 into zeroed U
    k_wmma_gemm<2, 4><<<(BB / 32) * (DCC / 64) * 8 / 8, 256, 0, stream>>>(
        pb, cbt, U, nullptr, BB, DCC, NN, 8);

    // 7) valid lengths + final assembly
    k_vl<<<(BB + 255) / 256, 256, 0, stream>>>(attn, vl, BB, LL);
    k_emit<<<BB * LL, 256, 0, stream>>>(text_emb, attn, U, invsum, vl,
                                        out_emb, out_mask, out_scalar, LL, DCC);
}
